// AttentionBlock_49125835931983
// MI455X (gfx1250) — compile-verified, matrix operand order checked by reading, not run on hardware
//
#include <hip/hip_runtime.h>

// ---------------------------------------------------------------------------
// CDNA5 (gfx1250) attention block: GroupNorm -> QKV -> softmax(QK^T/sqrt(C))V
// -> proj -> residual.  All GEMM stages use v_wmma_f32_16x16x32_bf16 (wave32).
// ---------------------------------------------------------------------------

typedef __attribute__((ext_vector_type(16))) __bf16 v16bf;
typedef __attribute__((ext_vector_type(8)))  float  v8f;

union FragAB { unsigned int u[8]; v16bf v; };
union FragCD { float f[8]; v8f v; };

#define BATCH 32
#define CH    256
#define NPIX  1024
#define NGRP  8
#define CPG   32           // channels per group

__device__ __forceinline__ unsigned short f2bf(float x) {
  unsigned int u = __float_as_uint(x);
  u += 0x7fffu + ((u >> 16) & 1u);           // round-to-nearest-even
  return (unsigned short)(u >> 16);
}

// Load a 16x32 bf16 A- or B-fragment.  `p` points to the *lane's* row (A) or
// column (B) start, with the K dimension contiguous in memory.  K-pair slot i
// in VGPR i follows the ISA 16-bit A-matrix table:
//   lanes 0-15 : V0..V3 -> K 0..7,  V4..V7 -> K 16..23
//   lanes 16-31: V0..V3 -> K 8..15, V4..V7 -> K 24..31
__device__ __forceinline__ v16bf load_ab_frag(const unsigned short* p) {
  const int half = (threadIdx.x >> 4) & 1;
  const unsigned int* p32 = (const unsigned int*)p;
  FragAB f;
#pragma unroll
  for (int i = 0; i < 8; ++i) {
    int kb = (i < 4) ? (half * 8 + 2 * i) : (16 + half * 8 + 2 * (i - 4));
    f.u[i] = p32[kb >> 1];
  }
  return f.v;
}

__device__ __forceinline__ v8f wmma_bf16(v16bf a, v16bf b, v8f c) {
  return __builtin_amdgcn_wmma_f32_16x16x32_bf16(
      /*neg_a=*/false, a, /*neg_b=*/false, b,
      /*c_mod=*/(short)0, c, /*reuse_a=*/false, /*reuse_b=*/false);
}

// ---------------------------------------------------------------------------
// Kernel 0: fp32 -> bf16 weight conversion (qkv_w: 768x256, proj_w: 256x256)
// ---------------------------------------------------------------------------
__global__ void k_cvt_weights(const float* __restrict__ qkv_w,
                              const float* __restrict__ proj_w,
                              unsigned short* __restrict__ wq_bf,
                              unsigned short* __restrict__ wp_bf) {
  int i = blockIdx.x * 256 + threadIdx.x;
  if (i < 3 * CH * CH) wq_bf[i] = f2bf(qkv_w[i]);
  if (i < CH * CH)     wp_bf[i] = f2bf(proj_w[i]);
}

// ---------------------------------------------------------------------------
// Kernel 1: GroupNorm.  One block per (batch, group).  Output h stored as
// bf16 in [B][N][C] layout (channel-contiguous per pixel) for WMMA B-frags.
// ---------------------------------------------------------------------------
__global__ void k_groupnorm(const float* __restrict__ x,
                            const float* __restrict__ nw,
                            const float* __restrict__ nb,
                            unsigned short* __restrict__ h_t) {
  const int bb = blockIdx.x >> 3;
  const int g  = blockIdx.x & 7;
  const int c0 = g * CPG;
  const float* xp = x + ((size_t)bb * CH + c0) * NPIX;   // 32*1024 contiguous

  float s = 0.f, s2 = 0.f;
  for (int e = threadIdx.x; e < CPG * NPIX; e += 256) {
    float v = xp[e];
    s += v; s2 += v * v;
  }
  __shared__ float rs[256], rs2[256];
  rs[threadIdx.x] = s; rs2[threadIdx.x] = s2;
  __syncthreads();
  for (int off = 128; off > 0; off >>= 1) {
    if (threadIdx.x < off) {
      rs[threadIdx.x]  += rs[threadIdx.x + off];
      rs2[threadIdx.x] += rs2[threadIdx.x + off];
    }
    __syncthreads();
  }
  const float inv_n = 1.0f / (float)(CPG * NPIX);
  const float mean  = rs[0] * inv_n;
  const float var   = rs2[0] * inv_n - mean * mean;
  const float rstd  = rsqrtf(var + 1e-5f);

  for (int e = threadIdx.x; e < CPG * NPIX; e += 256) {
    int c = c0 + (e >> 10);
    int n = e & 1023;
    float v = (xp[e] - mean) * rstd * nw[c] + nb[c];
    h_t[((size_t)bb * NPIX + n) * CH + c] = f2bf(v);
  }
}

// ---------------------------------------------------------------------------
// Kernel 2: QKV GEMM.  qkv[o,n] = sum_c W[o,c] * h[n,c] + b[o].
// Block covers 64 (o) x 128 (n); 8 waves, each wave a 32x32 tile (2x2 WMMA).
// Q,K stored [B][N][C] bf16; V stored [B][C][N] bf16.
// ---------------------------------------------------------------------------
__global__ void k_qkv(const unsigned short* __restrict__ wq_bf,
                      const float* __restrict__ qkv_b,
                      const unsigned short* __restrict__ h_t,
                      unsigned short* __restrict__ q_t,
                      unsigned short* __restrict__ k_t,
                      unsigned short* __restrict__ v_t) {
  const int bb   = blockIdx.z;
  const int n0   = blockIdx.x * 128 + (int)((threadIdx.x >> 5) & 3) * 32;
  const int o0   = blockIdx.y * 64  + (int)((threadIdx.x >> 7)) * 32;
  const int lane = threadIdx.x & 31;
  const int row  = lane & 15;
  const int half = lane >> 4;

  v8f acc00 = {}, acc01 = {}, acc10 = {}, acc11 = {};
  const unsigned short* hb = h_t + (size_t)bb * NPIX * CH;

#pragma unroll
  for (int k0 = 0; k0 < CH; k0 += 32) {
    v16bf a0 = load_ab_frag(wq_bf + (size_t)(o0 + row) * CH + k0);
    v16bf a1 = load_ab_frag(wq_bf + (size_t)(o0 + 16 + row) * CH + k0);
    v16bf b0 = load_ab_frag(hb + (size_t)(n0 + row) * CH + k0);
    v16bf b1 = load_ab_frag(hb + (size_t)(n0 + 16 + row) * CH + k0);
    acc00 = wmma_bf16(a0, b0, acc00);
    acc01 = wmma_bf16(a0, b1, acc01);
    acc10 = wmma_bf16(a1, b0, acc10);
    acc11 = wmma_bf16(a1, b1, acc11);
  }

  FragCD d[2][2];
  d[0][0].v = acc00; d[0][1].v = acc01; d[1][0].v = acc10; d[1][1].v = acc11;

#pragma unroll
  for (int mi = 0; mi < 2; ++mi) {
#pragma unroll
    for (int ni = 0; ni < 2; ++ni) {
      int nn = n0 + ni * 16 + row;
#pragma unroll
      for (int r = 0; r < 8; ++r) {
        int oo = o0 + mi * 16 + r + half * 8;
        float val = d[mi][ni].f[r] + qkv_b[oo];
        unsigned short bv = f2bf(val);
        int seg = oo >> 8, oc = oo & 255;
        if (seg == 0)
          q_t[((size_t)bb * NPIX + nn) * CH + oc] = bv;
        else if (seg == 1)
          k_t[((size_t)bb * NPIX + nn) * CH + oc] = bv;
        else
          v_t[((size_t)bb * CH + oc) * NPIX + nn] = bv;
      }
    }
  }
}

// ---------------------------------------------------------------------------
// Kernel 3: attention.  Block = (16 queries, 1 batch); 8 waves; wave w owns
// keys [w*128, w*128+128).  S kept in registers; cross-wave softmax via LDS;
// probs staged bf16 in LDS for the P*V WMMA pass.
// ---------------------------------------------------------------------------
__global__ void k_attn(const unsigned short* __restrict__ q_t,
                       const unsigned short* __restrict__ k_t,
                       const unsigned short* __restrict__ v_t,
                       unsigned short* __restrict__ hattn_t) {
  const int bb   = blockIdx.y;
  const int q0   = blockIdx.x * 16;
  const int wave = threadIdx.x >> 5;
  const int lane = threadIdx.x & 31;
  const int row  = lane & 15;
  const int half = lane >> 4;
  const float scale = 0.0625f;                 // 256^-0.5

  __shared__ unsigned short pbuf[16 * NPIX];   // 32 KB probs (bf16)
  __shared__ float red[16][128];               // 8 KB partial max/sum
  __shared__ float rowmax[16], rowsum[16];

  // Preload all 8 K-step A-fragments of the 16-query Q tile.
  v16bf qa[8];
  const unsigned short* qrow = q_t + ((size_t)bb * NPIX + q0 + row) * CH;
#pragma unroll
  for (int ks = 0; ks < 8; ++ks) qa[ks] = load_ab_frag(qrow + ks * 32);

  // S = Q K^T for this wave's 128 keys (8 tiles of 16 keys).
  FragCD s[8];
  const int kb0 = wave * 128;
#pragma unroll
  for (int kt = 0; kt < 8; ++kt) {
    v8f acc = {};
    const unsigned short* krow =
        k_t + ((size_t)bb * NPIX + kb0 + kt * 16 + row) * CH;
#pragma unroll
    for (int ks = 0; ks < 8; ++ks)
      acc = wmma_bf16(qa[ks], load_ab_frag(krow + ks * 32), acc);
    s[kt].v = acc;
#pragma unroll
    for (int r = 0; r < 8; ++r) s[kt].f[r] *= scale;
  }

  // Per-lane row maxima over this wave's keys -> LDS -> global row max.
  const int slot = wave * 16 + row;            // unique per (wave, key col)
#pragma unroll
  for (int r = 0; r < 8; ++r) {
    float m = -3.4e38f;
#pragma unroll
    for (int kt = 0; kt < 8; ++kt) m = fmaxf(m, s[kt].f[r]);
    red[half * 8 + r][slot] = m;
  }
  __syncthreads();
  if (threadIdx.x < 16) {
    float m = -3.4e38f;
    for (int i = 0; i < 128; ++i) m = fmaxf(m, red[threadIdx.x][i]);
    rowmax[threadIdx.x] = m;
  }
  __syncthreads();

  // exp in place, partial sums -> LDS -> global row sum.
#pragma unroll
  for (int r = 0; r < 8; ++r) {
    float mr = rowmax[half * 8 + r];
    float ps = 0.f;
#pragma unroll
    for (int kt = 0; kt < 8; ++kt) {
      float e = __expf(s[kt].f[r] - mr);
      s[kt].f[r] = e;
      ps += e;
    }
    red[half * 8 + r][slot] = ps;
  }
  __syncthreads();
  if (threadIdx.x < 16) {
    float su = 0.f;
    for (int i = 0; i < 128; ++i) su += red[threadIdx.x][i];
    rowsum[threadIdx.x] = su;
  }
  __syncthreads();

  // Normalized probs -> LDS as bf16, [query_row][key] contiguous.
#pragma unroll
  for (int r = 0; r < 8; ++r) {
    int m = half * 8 + r;
    float inv = 1.0f / rowsum[m];
#pragma unroll
    for (int kt = 0; kt < 8; ++kt) {
      int key = kb0 + kt * 16 + row;
      pbuf[m * NPIX + key] = f2bf(s[kt].f[r] * inv);
    }
  }
  __syncthreads();

  // P * V : wave handles 2 channel tiles; K-loop over all 1024 keys.
#pragma unroll
  for (int j = 0; j < 2; ++j) {
    const int c0 = (wave * 2 + j) * 16;
    v8f acc = {};
    const unsigned short* vrow = v_t + ((size_t)bb * CH + c0 + row) * NPIX;
    const unsigned short* prow = pbuf + row * NPIX;
#pragma unroll 4
    for (int ks = 0; ks < 32; ++ks) {
      v16bf pa = load_ab_frag(prow + ks * 32);
      v16bf vb = load_ab_frag(vrow + ks * 32);
      acc = wmma_bf16(pa, vb, acc);
    }
    FragCD dd; dd.v = acc;
#pragma unroll
    for (int r = 0; r < 8; ++r) {
      int qq = q0 + r + half * 8;
      int cc = c0 + row;
      hattn_t[((size_t)bb * NPIX + qq) * CH + cc] = f2bf(dd.f[r]);
    }
  }
}

// ---------------------------------------------------------------------------
// Kernel 4: proj GEMM + bias + residual.  out[b,o,n] = x + P[o,:] . hattn[n,:]
// Same tiling as k_qkv (block 64 x 128).
// ---------------------------------------------------------------------------
__global__ void k_proj(const unsigned short* __restrict__ wp_bf,
                       const float* __restrict__ proj_b,
                       const unsigned short* __restrict__ hattn_t,
                       const float* __restrict__ x,
                       float* __restrict__ out) {
  const int bb   = blockIdx.z;
  const int n0   = blockIdx.x * 128 + (int)((threadIdx.x >> 5) & 3) * 32;
  const int o0   = blockIdx.y * 64  + (int)((threadIdx.x >> 7)) * 32;
  const int lane = threadIdx.x & 31;
  const int row  = lane & 15;
  const int half = lane >> 4;

  v8f acc00 = {}, acc01 = {}, acc10 = {}, acc11 = {};
  const unsigned short* hb = hattn_t + (size_t)bb * NPIX * CH;

#pragma unroll
  for (int k0 = 0; k0 < CH; k0 += 32) {
    v16bf a0 = load_ab_frag(wp_bf + (size_t)(o0 + row) * CH + k0);
    v16bf a1 = load_ab_frag(wp_bf + (size_t)(o0 + 16 + row) * CH + k0);
    v16bf b0 = load_ab_frag(hb + (size_t)(n0 + row) * CH + k0);
    v16bf b1 = load_ab_frag(hb + (size_t)(n0 + 16 + row) * CH + k0);
    acc00 = wmma_bf16(a0, b0, acc00);
    acc01 = wmma_bf16(a0, b1, acc01);
    acc10 = wmma_bf16(a1, b0, acc10);
    acc11 = wmma_bf16(a1, b1, acc11);
  }

  FragCD d[2][2];
  d[0][0].v = acc00; d[0][1].v = acc01; d[1][0].v = acc10; d[1][1].v = acc11;

#pragma unroll
  for (int mi = 0; mi < 2; ++mi) {
#pragma unroll
    for (int ni = 0; ni < 2; ++ni) {
      int nn = n0 + ni * 16 + row;
#pragma unroll
      for (int r = 0; r < 8; ++r) {
        int oo = o0 + mi * 16 + r + half * 8;
        size_t idx = ((size_t)bb * CH + oo) * NPIX + nn;
        out[idx] = x[idx] + d[mi][ni].f[r] + proj_b[oo];
      }
    }
  }
}

// ---------------------------------------------------------------------------
// Host-side launcher.
// Inputs: 0:x 1:norm_w 2:norm_b 3:qkv_w 4:qkv_b 5:proj_w 6:proj_b
// ---------------------------------------------------------------------------
extern "C" void kernel_launch(void* const* d_in, const int* in_sizes, int n_in,
                              void* d_out, int out_size, void* d_ws, size_t ws_size,
                              hipStream_t stream) {
  (void)in_sizes; (void)n_in; (void)out_size; (void)ws_size;
  const float* x      = (const float*)d_in[0];
  const float* norm_w = (const float*)d_in[1];
  const float* norm_b = (const float*)d_in[2];
  const float* qkv_w  = (const float*)d_in[3];
  const float* qkv_b  = (const float*)d_in[4];
  const float* proj_w = (const float*)d_in[5];
  const float* proj_b = (const float*)d_in[6];
  float* out = (float*)d_out;

  char* ws = (char*)d_ws;
  const size_t SZ_WQ = (size_t)3 * CH * CH * 2;            // 384 KB
  const size_t SZ_WP = (size_t)CH * CH * 2;                // 128 KB
  const size_t SZ_T  = (size_t)BATCH * NPIX * CH * 2;      // 16 MB per tensor
  unsigned short* wq_bf   = (unsigned short*)(ws);
  unsigned short* wp_bf   = (unsigned short*)(ws + SZ_WQ);
  unsigned short* h_t     = (unsigned short*)(ws + SZ_WQ + SZ_WP);
  unsigned short* q_t     = (unsigned short*)(ws + SZ_WQ + SZ_WP + 1 * SZ_T);
  unsigned short* k_t     = (unsigned short*)(ws + SZ_WQ + SZ_WP + 2 * SZ_T);
  unsigned short* v_t     = (unsigned short*)(ws + SZ_WQ + SZ_WP + 3 * SZ_T);
  unsigned short* hattn_t = (unsigned short*)(ws + SZ_WQ + SZ_WP + 4 * SZ_T);

  k_cvt_weights<<<dim3((3 * CH * CH + 255) / 256), dim3(256), 0, stream>>>(
      qkv_w, proj_w, wq_bf, wp_bf);

  k_groupnorm<<<dim3(BATCH * NGRP), dim3(256), 0, stream>>>(
      x, norm_w, norm_b, h_t);

  k_qkv<<<dim3(NPIX / 128, (3 * CH) / 64, BATCH), dim3(256), 0, stream>>>(
      wq_bf, qkv_b, h_t, q_t, k_t, v_t);

  k_attn<<<dim3(NPIX / 16, BATCH), dim3(256), 0, stream>>>(
      q_t, k_t, v_t, hattn_t);

  k_proj<<<dim3(NPIX / 128, CH / 64, BATCH), dim3(256), 0, stream>>>(
      wp_bf, proj_b, hattn_t, x, out);
}
